// MinibatchDiscriminator_35957466202408
// MI455X (gfx1250) — matrix operand order repaired and checked by reading, not compiled, not run on hardware
//
#include <hip/hip_runtime.h>

#define BN     1024          // batch
#define INF    1024          // in features
#define OUTF   50
#define KD     8
#define NF     (OUTF * KD)   // 400 = columns of M
#define OCOLS  (INF + OUTF)  // 1074 output columns

typedef __attribute__((ext_vector_type(2)))  float    v2f;
typedef __attribute__((ext_vector_type(4)))  float    v4f;
typedef __attribute__((ext_vector_type(8)))  float    v8f;
typedef __attribute__((ext_vector_type(4)))  _Float16 v4h;
typedef __attribute__((ext_vector_type(8)))  _Float16 v8h;
typedef __attribute__((ext_vector_type(16))) _Float16 v16h;

// ---------------- convert x (f32 -> f16, row-major) ----------------
__global__ void __launch_bounds__(256)
cvt_x_kernel(const float* __restrict__ x, _Float16* __restrict__ xh) {
    int i = (blockIdx.x * 256 + threadIdx.x) * 4;       // 1M elements / 4
    v4f v = *(const v4f*)(x + i);
    v4h h;
    h.x = (_Float16)v.x; h.y = (_Float16)v.y;
    h.z = (_Float16)v.z; h.w = (_Float16)v.w;
    *(v4h*)(xh + i) = h;
}

// ------------- convert + transpose T: Tt[n*1024 + k] = T[k*400 + n] -------------
__global__ void __launch_bounds__(256)
cvt_T_kernel(const float* __restrict__ T, _Float16* __restrict__ Tt) {
    int idx = blockIdx.x * 256 + threadIdx.x;           // 400*1024 elements
    int k = idx & (INF - 1);
    int n = idx >> 10;
    Tt[idx] = (_Float16)T[k * NF + n];
}

// ---------------- WMMA GEMM: M[1024x400] = xh[1024x1024] * T ----------------
// One 16x16 C tile per wave; K-loop in steps of 32 with v_wmma_f32_16x16x32_f16.
__global__ void __launch_bounds__(256)
gemm_wmma_kernel(const _Float16* __restrict__ xh,
                 const _Float16* __restrict__ Tt,
                 float* __restrict__ M) {
    const int wave = threadIdx.x >> 5;
    const int lane = threadIdx.x & 31;
    const int tile = blockIdx.x * 8 + wave;             // 0..1599
    const int tm   = tile / 25;                         // 64 row tiles
    const int tn   = tile % 25;                         // 25 col tiles
    const int hi   = lane >> 4;                         // lane group 0/1
    const int l    = lane & 15;

    // A fragment rows / B fragment cols owned by this lane (per CDNA5 layout)
    const _Float16* Arow = xh + (size_t)(tm * 16 + l) * INF;
    const _Float16* Bcol = Tt + (size_t)(tn * 16 + l) * INF;

    v8f acc = {};
    for (int k0 = 0; k0 < INF; k0 += 32) {
        union { v16h v; struct { v8h lo, hi; } s; } a, b;
        // A 16x32 f16: lanes 0-15 hold K = k0+0..7 and k0+16..23 of row M=l;
        //              lanes 16-31 hold K = k0+8..15 and k0+24..31.
        a.s.lo = *(const v8h*)(Arow + k0 + hi * 8);
        a.s.hi = *(const v8h*)(Arow + k0 + 16 + hi * 8);
        // B 32x16 f16: lanes 0-15 hold K = k0+0..7 and k0+16..23 of col N=l;
        //              lanes 16-31 hold K = k0+8..15 and k0+24..31.
        b.s.lo = *(const v8h*)(Bcol + k0 + hi * 8);
        b.s.hi = *(const v8h*)(Bcol + k0 + 16 + hi * 8);
        acc = __builtin_amdgcn_wmma_f32_16x16x32_f16(
            /*neg_a=*/false, a.v, /*neg_b=*/false, b.v,
            /*c_mod=*/(short)0, acc, /*reuse_a=*/false, /*reuse_b=*/false);
    }

    // C/D layout: VGPR r -> row (tm*16 + hi*8 + r), col (tn*16 + l)
    float* Crow = M + (size_t)(tm * 16 + hi * 8) * NF + tn * 16 + l;
#pragma unroll
    for (int r = 0; r < 8; ++r) Crow[(size_t)r * NF] = acc[r];
}

// ------------- pairwise L1 + exp reduction -------------
// grid: (o = 0..49, jblock = 0..3), 256 threads; M[:, o, :] (32 KB) staged in LDS.
__global__ void __launch_bounds__(256)
pairwise_kernel(const float* __restrict__ M, float* __restrict__ out) {
    __shared__ float Ms[BN * KD];                       // 32 KB
    const int o   = blockIdx.x;
    const int tid = threadIdx.x;

    for (int i = tid; i < BN; i += 256) {
        const v4f* src = (const v4f*)(M + (size_t)i * NF + o * KD);
        *(v4f*)(Ms + i * KD)     = src[0];
        *(v4f*)(Ms + i * KD + 4) = src[1];
    }
    __syncthreads();

    const int j = blockIdx.y * 256 + tid;
    float mj[KD];
#pragma unroll
    for (int k = 0; k < KD; ++k) mj[k] = Ms[j * KD + k];

    float acc = 0.0f;
    for (int i = 0; i < BN; ++i) {
        float d = 0.0f;
#pragma unroll
        for (int k = 0; k < KD; ++k) d += fabsf(Ms[i * KD + k] - mj[k]);
        acc += __expf(-d);
    }
    out[(size_t)j * OCOLS + INF + o] = acc;
}

// ------------- copy x into out[:, 0:1024] (8B-aligned vectors) -------------
__global__ void __launch_bounds__(256)
copy_x_kernel(const float* __restrict__ x, float* __restrict__ out) {
    int idx = blockIdx.x * 256 + threadIdx.x;           // 1024 * 512 float2's
    int j = idx >> 9;
    int c = (idx & 511) * 2;
    *(v2f*)(out + (size_t)j * OCOLS + c) = *(const v2f*)(x + (size_t)j * INF + c);
}

extern "C" void kernel_launch(void* const* d_in, const int* in_sizes, int n_in,
                              void* d_out, int out_size, void* d_ws, size_t ws_size,
                              hipStream_t stream) {
    const float* x = (const float*)d_in[0];   // [1024, 1024] f32
    const float* T = (const float*)d_in[1];   // [1024, 400]  f32
    float* out = (float*)d_out;               // [1024, 1074] f32

    char* ws = (char*)d_ws;
    _Float16* xh = (_Float16*)ws;                                   // 2 MB
    _Float16* Tt = (_Float16*)(ws + (size_t)BN * INF * 2);          // 0.8 MB
    float*    M  = (float*)   (ws + (size_t)BN * INF * 2
                                  + (size_t)NF * INF * 2);          // 1.6 MB

    // x -> f16 (1M elems, 4/thread)
    cvt_x_kernel<<<(BN * INF) / (256 * 4), 256, 0, stream>>>(x, xh);
    // T -> f16 transposed (400K elems)
    cvt_T_kernel<<<(NF * INF) / 256, 256, 0, stream>>>(T, Tt);
    // GEMM: 64x25 = 1600 tiles, 8 waves (tiles) per block
    gemm_wmma_kernel<<<1600 / 8, 256, 0, stream>>>(xh, Tt, M);
    // pairwise exp-kernel reduction: 50 outs x 4 j-slabs
    pairwise_kernel<<<dim3(OUTF, BN / 256), 256, 0, stream>>>(M, out);
    // passthrough concat of x
    copy_x_kernel<<<(BN * (INF / 2)) / 256, 256, 0, stream>>>(x, out);
}
